// Transformer_1108101562972
// MI455X (gfx1250) — matrix operand order repaired and checked by reading, not compiled, not run on hardware
//
#include <hip/hip_runtime.h>
#include <hip/hip_bf16.h>
#include <math.h>

typedef __attribute__((ext_vector_type(16))) _Float16 v16h;
typedef __attribute__((ext_vector_type(8)))  float    v8f;
typedef __attribute__((ext_vector_type(4)))  unsigned int u32x4;
typedef __attribute__((ext_vector_type(8)))  int      i32x8;
typedef __attribute__((ext_vector_type(4)))  int      i32x4;

#define WMMA_F16(a, b, c) \
  __builtin_amdgcn_wmma_f32_16x16x32_f16(false, (a), false, (b), (short)0, (c), false, false)

#if __has_builtin(__builtin_amdgcn_tensor_load_to_lds) && \
    __has_builtin(__builtin_amdgcn_s_wait_tensorcnt)
#define HAVE_TDM 1
#else
#define HAVE_TDM 0
#endif

// ---------------------------------------------------------------------------
// Fragment load helpers (LDS or global). Both use contiguous K (k = 16*(lane/16) + s), not the ISA table order.
// A-frag (16x32 f16): lane<16 -> K 0..15, lane>=16 -> K 16..31 (contiguous, same as B)
//   => one contiguous 16xf16 (32B) run at offset fo = (lane<16)?0:16 (ld_frag16).
// B-frag (32x16 f16): lane<16 -> K 0..15, lane>=16 -> K 16..31 (contiguous 16 f16).
// ---------------------------------------------------------------------------
__device__ inline v16h ld_frag2x8(const _Float16* p) {
  v16h r;
  *reinterpret_cast<float4*>(&r)       = *reinterpret_cast<const float4*>(p);
  *(reinterpret_cast<float4*>(&r) + 1) = *reinterpret_cast<const float4*>(p + 16);
  return r;
}
__device__ inline v16h ld_frag16(const _Float16* p) {
  v16h r;
  *reinterpret_cast<float4*>(&r)       = *reinterpret_cast<const float4*>(p);
  *(reinterpret_cast<float4*>(&r) + 1) = *reinterpret_cast<const float4*>(p + 8);
  return r;
}

__device__ inline float xor_max16(float x) {
#pragma unroll
  for (int m = 1; m < 16; m <<= 1) x = fmaxf(x, __shfl_xor(x, m, 32));
  return x;
}
__device__ inline float xor_sum16(float x) {
#pragma unroll
  for (int m = 1; m < 16; m <<= 1) x += __shfl_xor(x, m, 32);
  return x;
}

#define LDT 40  // f16 row stride in LDS (80B = 20 DWORDs: 16B-aligned, conflict-free)

// ---------------------------------------------------------------------------
// TDM: DMA one [128 rows x 32 f16] tile (row stride K f16 in global) into LDS.
// LDS padding via D# pad engine: 16 DWORDs data (pad_interval=3) + 4 DWORDs pad
// (pad_amount=3) per row => 20-DWORD rows == LDT f16 stride.
// D# group0: count=1 | lds_addr | global_addr | type=2.
// D# group1: data_size=1(2B), pad_enable, tensor_dim0=K, tensor_dim1=rows,
//            tile_dim0=32, tile_dim1=128, tensor_dim0_stride=K.
// Wave-level op (EXEC ignored), tracked by TENSORcnt.
// This toolchain uses the 6-arg builtin: (g0, g1, g2, g3, g4, cpol).
// ---------------------------------------------------------------------------
#if HAVE_TDM
__device__ inline void tdm_load_tile128x32(unsigned int lds_addr,
                                           const _Float16* gptr,
                                           int K, int rows) {
  const unsigned long long ga = (unsigned long long)(uintptr_t)gptr;
  u32x4 g0;
  g0[0] = 1u;                                   // count=1, user mode, no gather
  g0[1] = lds_addr;                             // LDS byte address
  g0[2] = (unsigned int)ga;                     // global_addr[31:0]
  g0[3] = (unsigned int)((ga >> 32) & 0x01FFFFFFu) | 0x80000000u; // [56:32]|type=2
  i32x8 g1;
  g1[0] = (1 << 16) | (1 << 20) | (3 << 22) | (3 << 25); // dsz=2B, pad 16+4 dw
  g1[1] = (K & 0xFFFF) << 16;                   // tensor_dim0 low16
  g1[2] = ((unsigned)K >> 16) | ((rows & 0xFFFF) << 16); // dim0 hi / dim1 lo
  g1[3] = ((unsigned)rows >> 16) | (32 << 16);  // dim1 hi / tile_dim0=32
  g1[4] = 128;                                  // tile_dim1=128, tile_dim2=0
  g1[5] = K;                                    // tensor_dim0_stride low32
  g1[6] = 0;
  g1[7] = 0;
  const i32x4 z4 = {0, 0, 0, 0};
  const i32x8 z8 = {0, 0, 0, 0, 0, 0, 0, 0};
  __builtin_amdgcn_tensor_load_to_lds(g0, g1, z4, z4, z8, 0);
}
#endif

// ---------------------------------------------------------------------------
// LayerNorm over D=1024, emit f16 row-major. One row per 256-thread block.
// ---------------------------------------------------------------------------
__global__ __launch_bounds__(256) void ln_f16_kernel(
    const float* __restrict__ x, const float* __restrict__ scale,
    const float* __restrict__ bias, _Float16* __restrict__ out) {
  const int row = blockIdx.x;
  const float* xr = x + (size_t)row * 1024;
  const int t = threadIdx.x;
  float v[4];
#pragma unroll
  for (int i = 0; i < 4; i++) v[i] = xr[t + i * 256];
  float s = 0.f, s2 = 0.f;
#pragma unroll
  for (int i = 0; i < 4; i++) { s += v[i]; s2 += v[i] * v[i]; }
#pragma unroll
  for (int m = 1; m < 32; m <<= 1) { s += __shfl_xor(s, m, 32); s2 += __shfl_xor(s2, m, 32); }
  __shared__ float ss[8], ss2[8];
  if ((t & 31) == 0) { ss[t >> 5] = s; ss2[t >> 5] = s2; }
  __syncthreads();
  s = 0.f; s2 = 0.f;
#pragma unroll
  for (int i = 0; i < 8; i++) { s += ss[i]; s2 += ss2[i]; }
  const float mean = s * (1.f / 1024.f);
  const float var  = s2 * (1.f / 1024.f) - mean * mean;
  const float r    = rsqrtf(var + 1e-5f);
  _Float16* orow = out + (size_t)row * 1024;
#pragma unroll
  for (int i = 0; i < 4; i++) {
    const int c = t + i * 256;
    orow[c] = (_Float16)((v[i] - mean) * r * scale[c] + bias[c]);
  }
}

// ---------------------------------------------------------------------------
// Transpose + f32->f16 convert: in [K,N] row-major -> out [N,K] row-major.
// ---------------------------------------------------------------------------
__global__ __launch_bounds__(256) void transpose_cvt(
    const float* __restrict__ in, _Float16* __restrict__ out, int K, int N) {
  __shared__ float tile[32][33];
  const int nb = blockIdx.x * 32, kb = blockIdx.y * 32;
  const int tx = threadIdx.x, ty = threadIdx.y;
#pragma unroll
  for (int j = 0; j < 32; j += 8)
    tile[ty + j][tx] = in[(size_t)(kb + ty + j) * N + nb + tx];
  __syncthreads();
#pragma unroll
  for (int j = 0; j < 32; j += 8)
    out[(size_t)(nb + ty + j) * K + kb + tx] = (_Float16)tile[tx][ty + j];
}

// ---------------------------------------------------------------------------
// WMMA GEMM: C[M,N] (f32 acc) = A16[M,K] @ BT16[N,K]^T.
// Block tile 128x128, 8 waves of 32x64, BK=32.
// Tiles staged by TDM (double-buffered, TENSORcnt-paced) when available.
// MODE 0: Cf = acc                  (QKV projection)
// MODE 1: Cf = Cf + bias + acc      (residual in-place: out-proj / FFN2)
// MODE 2: Ch = f16(gelu(acc+bias))  (FFN1 -> feeds FFN2 directly)
// ---------------------------------------------------------------------------
template <int MODE>
__global__ __launch_bounds__(256) void gemm_wmma(
    const _Float16* __restrict__ A, const _Float16* __restrict__ BT,
    float* __restrict__ Cf, _Float16* __restrict__ Ch,
    const float* __restrict__ bias, int M, int N, int K) {
  __shared__ __align__(16) _Float16 As[2][128 * LDT];
  __shared__ __align__(16) _Float16 Bs[2][128 * LDT];
  const int m0 = blockIdx.y * 128, n0 = blockIdx.x * 128;
  const int t = threadIdx.x, wave = t >> 5, lane = t & 31;
  const int wm = (wave & 3) * 32;   // 4 waves tile M (4*32 = 128)
  const int wn = (wave >> 2) * 64;  // 2 waves tile N (2*64 = 128)
  const int fr = lane & 15;
  const int foA = (lane < 16) ? 0 : 16;
  const int foB = (lane < 16) ? 0 : 16;

  v8f acc[2][4];
#pragma unroll
  for (int i = 0; i < 2; i++)
#pragma unroll
    for (int j = 0; j < 4; j++)
#pragma unroll
      for (int e = 0; e < 8; e++) acc[i][j][e] = 0.f;

  const _Float16* Abase = A + (size_t)m0 * K;
  const _Float16* Bbase = BT + (size_t)n0 * K;
  const int nTiles = K >> 5;

#if HAVE_TDM
  const unsigned int ldsA0 = (unsigned int)(uintptr_t)&As[0][0];
  const unsigned int ldsA1 = (unsigned int)(uintptr_t)&As[1][0];
  const unsigned int ldsB0 = (unsigned int)(uintptr_t)&Bs[0][0];
  const unsigned int ldsB1 = (unsigned int)(uintptr_t)&Bs[1][0];
  if (wave == 0) {  // prime buffer 0
    tdm_load_tile128x32(ldsA0, Abase, K, M);
    tdm_load_tile128x32(ldsB0, Bbase, K, N);
  }
#else
  const int lr = t >> 1, lk = (t & 1) * 16;
  const _Float16* Ag = Abase + (size_t)lr * K + lk;
  const _Float16* Bg = Bbase + (size_t)lr * K + lk;
  _Float16* AsW0 = &As[0][0] + lr * LDT + lk;
  _Float16* BsW0 = &Bs[0][0] + lr * LDT + lk;
#endif

  for (int i = 0; i < nTiles; i++) {
    const int buf = i & 1;
#if HAVE_TDM
    if (wave == 0) {
      if (i + 1 < nTiles) {  // prefetch next tile into the other buffer
        tdm_load_tile128x32(buf ? ldsA0 : ldsA1, Abase + (i + 1) * 32, K, M);
        tdm_load_tile128x32(buf ? ldsB0 : ldsB1, Bbase + (i + 1) * 32, K, N);
        __builtin_amdgcn_s_wait_tensorcnt(2);  // current tile done (in-order)
      } else {
        __builtin_amdgcn_s_wait_tensorcnt(0);
      }
    }
    __syncthreads();  // publish TDM-written LDS to all waves
#else
    {
      _Float16* aw = AsW0 + buf * 128 * LDT;
      _Float16* bw = BsW0 + buf * 128 * LDT;
      const _Float16* ag = Ag + i * 32;
      const _Float16* bg = Bg + i * 32;
      reinterpret_cast<float4*>(aw)[0] = reinterpret_cast<const float4*>(ag)[0];
      reinterpret_cast<float4*>(aw)[1] = reinterpret_cast<const float4*>(ag)[1];
      reinterpret_cast<float4*>(bw)[0] = reinterpret_cast<const float4*>(bg)[0];
      reinterpret_cast<float4*>(bw)[1] = reinterpret_cast<const float4*>(bg)[1];
    }
    __syncthreads();
#endif

    const _Float16* as = &As[buf][0];
    const _Float16* bs = &Bs[buf][0];
    v16h af[2], bf[4];
#pragma unroll
    for (int mt = 0; mt < 2; mt++)
      af[mt] = ld_frag16(as + (wm + mt * 16 + fr) * LDT + foA);
#pragma unroll
    for (int nt = 0; nt < 4; nt++)
      bf[nt] = ld_frag16(bs + (wn + nt * 16 + fr) * LDT + foB);
#pragma unroll
    for (int nt = 0; nt < 4; nt++)
#pragma unroll
      for (int mt = 0; mt < 2; mt++)
        acc[mt][nt] = WMMA_F16(af[mt], bf[nt], acc[mt][nt]);
    __syncthreads();  // all waves done reading buf before it is refilled
  }

  // epilogue: C layout -> row = v + 8*(lane>>4), col = lane&15
  const int rhalf = (lane >> 4) * 8;
#pragma unroll
  for (int mt = 0; mt < 2; mt++)
#pragma unroll
    for (int nt = 0; nt < 4; nt++)
#pragma unroll
      for (int v = 0; v < 8; v++) {
        const int gm = m0 + wm + mt * 16 + v + rhalf;
        const int gn = n0 + wn + nt * 16 + (lane & 15);
        const float val = acc[mt][nt][v];
        if (MODE == 0) {
          Cf[(size_t)gm * N + gn] = val;
        } else if (MODE == 1) {
          Cf[(size_t)gm * N + gn] = Cf[(size_t)gm * N + gn] + bias[gn] + val;
        } else {
          const float u = val + bias[gn];
          Ch[(size_t)gm * N + gn] =
              (_Float16)(u * 0.5f * (1.f + erff(u * 0.70710678118f)));
        }
      }
  (void)M;
}

// ---------------------------------------------------------------------------
// Split QKV f32 [2048, 3072] -> f16 [B,H,N,DH]; scale Q by DH^-0.5 = 0.125.
// ---------------------------------------------------------------------------
__global__ __launch_bounds__(256) void pack_qkv(
    const float* __restrict__ C, _Float16* __restrict__ q,
    _Float16* __restrict__ k, _Float16* __restrict__ v) {
  const size_t id = (size_t)blockIdx.x * 256 + threadIdx.x;
  const int m = (int)(id / 3072), j = (int)(id % 3072);
  const int which = j >> 10, jj = j & 1023;
  const int h = jj >> 6, dh = jj & 63;
  const int b = m >> 9, n = m & 511;
  const size_t o = ((((size_t)b * 16 + h) * 512) + n) * 64 + dh;
  const float val = C[(size_t)m * 3072 + j];
  if (which == 0)      q[o] = (_Float16)(val * 0.125f);
  else if (which == 1) k[o] = (_Float16)val;
  else                 v[o] = (_Float16)val;
}

// ---------------------------------------------------------------------------
// Flash attention, one (b,h) x 128 query rows per block. Wave owns 16 rows.
// S = Q@K^T (2 WMMA/16-key tile), online softmax, O += P@V (4 WMMA/chunk).
// ---------------------------------------------------------------------------
__global__ __launch_bounds__(256) void attn_wmma(
    const _Float16* __restrict__ Q, const _Float16* __restrict__ Km,
    const _Float16* __restrict__ V, _Float16* __restrict__ O) {
  __shared__ __align__(16) _Float16 Ks[32 * 72];   // [key][dh], stride 72
  __shared__ __align__(16) _Float16 Vs[64 * LDT];  // [dh][key], stride 40
  __shared__ __align__(16) _Float16 Ps[8][16 * LDT];

  const int bid = blockIdx.x;
  const int bh = bid >> 2;             // b*16 + h
  const int q0 = (bid & 3) * 128;
  const int b = bh >> 4, h = bh & 15;
  const _Float16* Qb = Q  + (size_t)bh * 512 * 64;
  const _Float16* Kb = Km + (size_t)bh * 512 * 64;
  const _Float16* Vb = V  + (size_t)bh * 512 * 64;

  const int t = threadIdx.x, wave = t >> 5, lane = t & 31;
  const int foA = (lane < 16) ? 0 : 16;
  const int foB = (lane < 16) ? 0 : 16;
  const int qrow = q0 + wave * 16 + (lane & 15);

  // Q fragments pinned in registers (DH=64 -> two 16x32 A-frags)
  v16h qa[2];
#pragma unroll
  for (int kc = 0; kc < 2; kc++)
    qa[kc] = ld_frag16(Qb + (size_t)qrow * 64 + kc * 32 + foA);

  v8f oacc[4];
#pragma unroll
  for (int d = 0; d < 4; d++)
#pragma unroll
    for (int e = 0; e < 8; e++) oacc[d][e] = 0.f;
  float mrow[8], lrow[8];
#pragma unroll
  for (int v = 0; v < 8; v++) { mrow[v] = -1e30f; lrow[v] = 0.f; }

  for (int kt = 0; kt < 512; kt += 32) {
    __syncthreads();
    for (int i = t; i < 32 * 64; i += 256) {
      const int key = i >> 6, dh = i & 63;
      Ks[key * 72 + dh]  = Kb[(size_t)(kt + key) * 64 + dh];
      Vs[dh * LDT + key] = Vb[(size_t)(kt + key) * 64 + dh];  // transposed
    }
    __syncthreads();

    // S tiles (16 rows x two 16-key tiles)
    v8f s[2];
#pragma unroll
    for (int st = 0; st < 2; st++) {
      v8f c;
#pragma unroll
      for (int e = 0; e < 8; e++) c[e] = 0.f;
#pragma unroll
      for (int kc = 0; kc < 2; kc++) {
        const v16h kf = ld_frag16(Ks + (st * 16 + (lane & 15)) * 72 + kc * 32 + foB);
        c = WMMA_F16(qa[kc], kf, c);
      }
      s[st] = c;
    }

    // online softmax (row = v + 8*(lane>>4); 16 cols live in 16 lanes)
    float scl[8];
#pragma unroll
    for (int v = 0; v < 8; v++) {
      const float rmax = xor_max16(fmaxf(s[0][v], s[1][v]));
      const float mnew = fmaxf(mrow[v], rmax);
      scl[v] = __expf(mrow[v] - mnew);
      mrow[v] = mnew;
      const float p0 = __expf(s[0][v] - mnew);
      const float p1 = __expf(s[1][v] - mnew);
      const int prow = v + ((lane >> 4) * 8);
      Ps[wave][prow * LDT + (lane & 15)]      = (_Float16)p0;
      Ps[wave][prow * LDT + 16 + (lane & 15)] = (_Float16)p1;
      lrow[v] = lrow[v] * scl[v] + xor_sum16(p0 + p1);
    }
#pragma unroll
    for (int d = 0; d < 4; d++)
#pragma unroll
      for (int v = 0; v < 8; v++) oacc[d][v] *= scl[v];

    // O += P @ V_chunk  (per-wave Ps; same-wave LDS ops are ordered)
    const v16h pa = ld_frag16(&Ps[wave][(lane & 15) * LDT + foA]);
#pragma unroll
    for (int d = 0; d < 4; d++) {
      const v16h vf = ld_frag16(Vs + (d * 16 + (lane & 15)) * LDT + foB);
      oacc[d] = WMMA_F16(pa, vf, oacc[d]);
    }
  }

  // write O f16 [B, N, H*DH] (row-major GEMM-A layout for out-projection)
  const int rhalf = (lane >> 4) * 8;
#pragma unroll
  for (int d = 0; d < 4; d++)
#pragma unroll
    for (int v = 0; v < 8; v++) {
      const int qr = q0 + wave * 16 + v + rhalf;
      O[(((size_t)b * 512 + qr) * 1024) + h * 64 + d * 16 + (lane & 15)] =
          (_Float16)(oacc[d][v] / lrow[v]);
    }
}

// ---------------------------------------------------------------------------
// Host side: sequence the 6 layers. Workspace layout (bytes):
//   X    @ 0    : f32 [2048,1024]  (8 MB, running activations)
//   H16  @ 8M   : f16 [2048,1024]  (4 MB, LN output)
//   W16T @ 12M  : f16 weights^T    (8 MB, per-GEMM)
//   C    @ 20M  : f32 [2048,3072]  (24 MB, QKV out; reused as f16 GELU buf)
//   Q16  @ 44M, K16 @ 48M, V16 @ 52M : f16 [B,H,N,DH] (4 MB each)
//   O16  @ 56M  : f16 [2048,1024]  (4 MB)   -> total 60 MB
// ---------------------------------------------------------------------------
extern "C" void kernel_launch(void* const* d_in, const int* in_sizes, int n_in,
                              void* d_out, int out_size, void* d_ws, size_t ws_size,
                              hipStream_t stream) {
  const float* x    = (const float*)d_in[0];
  const float* ln1s = (const float*)d_in[1];
  const float* ln1b = (const float*)d_in[2];
  const float* wqkv = (const float*)d_in[3];
  const float* wout = (const float*)d_in[4];
  const float* bout = (const float*)d_in[5];
  const float* ln2s = (const float*)d_in[6];
  const float* ln2b = (const float*)d_in[7];
  const float* w1   = (const float*)d_in[8];
  const float* b1   = (const float*)d_in[9];
  const float* w2   = (const float*)d_in[10];
  const float* b2   = (const float*)d_in[11];
  (void)in_sizes; (void)n_in; (void)out_size; (void)ws_size;

  char* ws = (char*)d_ws;
  float*     X    = (float*)(ws);
  _Float16*  H16  = (_Float16*)(ws + ((size_t)8  << 20));
  _Float16*  W16T = (_Float16*)(ws + ((size_t)12 << 20));
  float*     C    = (float*)(ws + ((size_t)20 << 20));
  _Float16*  G16  = (_Float16*)C;  // FFN1 gelu output [2048,4096] f16 (16 MB)
  _Float16*  Q16  = (_Float16*)(ws + ((size_t)44 << 20));
  _Float16*  K16  = (_Float16*)(ws + ((size_t)48 << 20));
  _Float16*  V16  = (_Float16*)(ws + ((size_t)52 << 20));
  _Float16*  O16  = (_Float16*)(ws + ((size_t)56 << 20));

  const size_t actBytes = (size_t)2048 * 1024 * sizeof(float);
  hipError_t err = hipMemcpyAsync(X, x, actBytes, hipMemcpyDeviceToDevice, stream);
  (void)err;

  const dim3 tb(32, 8);
  for (int l = 0; l < 6; l++) {
    // --- attention block ---
    ln_f16_kernel<<<2048, 256, 0, stream>>>(X, ln1s + l * 1024, ln1b + l * 1024, H16);
    transpose_cvt<<<dim3(96, 32), tb, 0, stream>>>(wqkv + (size_t)l * 1024 * 3072, W16T, 1024, 3072);
    gemm_wmma<0><<<dim3(24, 16), 256, 0, stream>>>(H16, W16T, C, nullptr, nullptr, 2048, 3072, 1024);
    pack_qkv<<<24576, 256, 0, stream>>>(C, Q16, K16, V16);
    attn_wmma<<<256, 256, 0, stream>>>(Q16, K16, V16, O16);
    transpose_cvt<<<dim3(32, 32), tb, 0, stream>>>(wout + (size_t)l * 1024 * 1024, W16T, 1024, 1024);
    gemm_wmma<1><<<dim3(8, 16), 256, 0, stream>>>(O16, W16T, X, nullptr, bout + l * 1024, 2048, 1024, 1024);
    // --- feed-forward block ---
    ln_f16_kernel<<<2048, 256, 0, stream>>>(X, ln2s + l * 1024, ln2b + l * 1024, H16);
    transpose_cvt<<<dim3(128, 32), tb, 0, stream>>>(w1 + (size_t)l * 1024 * 4096, W16T, 1024, 4096);
    gemm_wmma<2><<<dim3(32, 16), 256, 0, stream>>>(H16, W16T, nullptr, G16, b1 + l * 4096, 2048, 4096, 1024);
    transpose_cvt<<<dim3(32, 128), tb, 0, stream>>>(w2 + (size_t)l * 4096 * 1024, W16T, 4096, 1024);
    gemm_wmma<1><<<dim3(8, 16), 256, 0, stream>>>(G16, W16T, X, nullptr, b2 + l * 1024, 2048, 1024, 4096);
  }
  err = hipMemcpyAsync(d_out, X, actBytes, hipMemcpyDeviceToDevice, stream);
  (void)err;
}
